// Model_68049461837949
// MI455X (gfx1250) — matrix operand order repaired and checked by reading, not compiled, not run on hardware
//
#include <hip/hip_runtime.h>
#include <math.h>

// ---------------------------------------------------------------------------
// ns-Informer forward for MI455X (gfx1250, wave32, WMMA 16x16x32 f16/f32-acc)
// B=8 S=336 N=8 D=512 H=8 Dh=64 D_FF=2048 P=96 E_LAYERS=2 HID=128 U=30
// ---------------------------------------------------------------------------

typedef _Float16 half_t;
typedef __attribute__((ext_vector_type(16))) _Float16 v16h;
typedef __attribute__((ext_vector_type(8)))  _Float16 v8h;
typedef __attribute__((ext_vector_type(8)))  float    v8f;

#define Bc 8
#define Sc 336
#define Nc 8
#define Dc 512
#define Hc 8
#define DHc 64
#define DFFc 2048
#define Pc 96
#define HIDc 128
#define Uc 30
#define BNc 64            // B*N
#define Mrows 21504       // BN*S

// GEMM epilogue mode bits
#define MB_BIAS 1
#define MB_GELU 2
#define MB_F16  4

__device__ inline v8f wmma_f16(v16h a, v16h b, v8f c) {
  // D = A(16x32,f16) x B(32x16,f16) + C(16x16,f32)
  return __builtin_amdgcn_wmma_f32_16x16x32_f16(false, a, false, b, (short)0, c,
                                                false, false);
}

// A fragment from row-major f16 tile (16 rows x >=32 k), stride in halfs.
// lane&15 = row, kb=(lane>>4)*16 ; halfs j -> K=kb+j
// Requires 16B alignment of tile rows (stride*2 % 16 == 0).
__device__ inline v16h load_afrag(const half_t* tile, int stride, int lane) {
  const half_t* rp = tile + (size_t)(lane & 15) * stride + ((lane >> 4) * 16);
  v8h x0 = *(const v8h*)rp;         // K = kb .. kb+7
  v8h x1 = *(const v8h*)(rp + 8);   // K = kb+8 .. kb+15
  v16h a;
#pragma unroll
  for (int j = 0; j < 8; ++j) { a[j] = x0[j]; a[8 + j] = x1[j]; }
  return a;
}

// B fragment straight from global Bt[N,K] f16: lane holds col n=colbase+(lane&15),
// halfs j -> K = k0 + (lane>>4)*16 + j  (16 contiguous halves -> 2x b128).
__device__ inline v16h load_bfrag_g(const half_t* Bt, int col, size_t K, int k0,
                                    int lane) {
  const half_t* bp = Bt + (size_t)col * K + k0 + (lane >> 4) * 16;
  v8h t0 = *(const v8h*)bp;
  v8h t1 = *(const v8h*)(bp + 8);
  v16h b;
#pragma unroll
  for (int j = 0; j < 8; ++j) { b[j] = t0[j]; b[8 + j] = t1[j]; }
  return b;
}

__device__ inline float gelu_exact(float x) {
  return 0.5f * x * (1.0f + erff(x * 0.70710678118654752f));
}

// ---------------------------------------------------------------------------
// GEMM: C[M,N] = A16[M,K] @ Bt16[N,K]^T (+bias (+GELU)), f16 WMMA, f32 acc.
// 256 threads = 8 waves; block tile 128x128; wave tile 64x32 (4x2 frags,
// 8 wmma per wave per 32-wide K-step). B read directly from global (L2-hot).
// ---------------------------------------------------------------------------
__global__ __launch_bounds__(256) void gemm_wmma16(
    const half_t* __restrict__ A, const half_t* __restrict__ Bt,
    const float* __restrict__ bias, void* __restrict__ Cout,
    int M, int N, int K, int mode) {
  __shared__ __attribute__((aligned(16))) half_t As[128][40];  // [m][k], pad 8
  const int tid = threadIdx.x;
  const int lane = tid & 31;
  const int wave = tid >> 5;
  const int m0 = blockIdx.y * 128;
  const int n0 = blockIdx.x * 128;
  const int wm = (wave & 1) * 64;   // wave row group
  const int wn = (wave >> 1) * 32;  // wave col group

  v8f vzero = {};
  v8f acc[4][2];
#pragma unroll
  for (int mi = 0; mi < 4; ++mi) { acc[mi][0] = vzero; acc[mi][1] = vzero; }

  const int r = tid >> 1;          // staging row (0..127)
  const int seg = (tid & 1) * 16;  // staging k-segment
  const int gm = m0 + r;
  const half_t* Arow = A + (size_t)gm * K + seg;

  // clamped B columns (results for clamped cols are discarded at store)
  int bcol0 = n0 + wn + (lane & 15);
  int bcol1 = bcol0 + 16;
  bcol0 = (bcol0 < N) ? bcol0 : 0;
  bcol1 = (bcol1 < N) ? bcol1 : 0;

  for (int k0 = 0; k0 < K; k0 += 32) {
    // stage A tile (128x32 halfs): 2x 16B loads + 2x 16B LDS stores per thread
    {
      v8h x0 = {}, x1 = {};
      if (gm < M) {
        const half_t* src = Arow + k0;
        x0 = *(const v8h*)src;
        x1 = *(const v8h*)(src + 8);
        if (k0 + 32 < K) __builtin_prefetch((const void*)(src + 32), 0, 1);
      }
      *(v8h*)&As[r][seg] = x0;
      *(v8h*)&As[r][seg + 8] = x1;
    }
    __syncthreads();
    v16h bf0 = load_bfrag_g(Bt, bcol0, (size_t)K, k0, lane);
    v16h bf1 = load_bfrag_g(Bt, bcol1, (size_t)K, k0, lane);
#pragma unroll
    for (int mi = 0; mi < 4; ++mi) {
      v16h af = load_afrag(&As[wm + mi * 16][0], 40, lane);
      acc[mi][0] = wmma_f16(af, bf0, acc[mi][0]);
      acc[mi][1] = wmma_f16(af, bf1, acc[mi][1]);
    }
    __syncthreads();
  }

  const int rb = (lane >> 4) * 8;
  const int cl = lane & 15;
#pragma unroll
  for (int mi = 0; mi < 4; ++mi)
#pragma unroll
    for (int ni = 0; ni < 2; ++ni) {
      int cb = n0 + wn + ni * 16 + cl;
      if (cb >= N) continue;
      float bval = (mode & MB_BIAS) ? bias[cb] : 0.f;
      int rbase = m0 + wm + mi * 16 + rb;
#pragma unroll
      for (int rr = 0; rr < 8; ++rr) {
        int gr = rbase + rr;
        if (gr >= M) continue;
        float v = acc[mi][ni][rr] + bval;
        if (mode & MB_GELU) v = gelu_exact(v);
        if (mode & MB_F16) ((half_t*)Cout)[(size_t)gr * N + cb] = (half_t)v;
        else               ((float*)Cout)[(size_t)gr * N + cb] = v;
      }
    }
}

// Tiled transpose + f32->f16 convert: Bt[n*K+k] = (half)B[k*N+n]
__global__ __launch_bounds__(256) void k_transpose_cvt(
    const float* __restrict__ B, half_t* __restrict__ Bt, int K, int N) {
  __shared__ float t[32][33];
  int k0 = blockIdx.x * 32, n0 = blockIdx.y * 32;
  int lx = threadIdx.x & 31, ly = threadIdx.x >> 5;
  for (int yy = ly; yy < 32; yy += 8) {
    int k = k0 + yy, n = n0 + lx;
    t[yy][lx] = (k < K && n < N) ? B[(size_t)k * N + n] : 0.f;
  }
  __syncthreads();
  for (int yy = ly; yy < 32; yy += 8) {
    int n = n0 + yy, k = k0 + lx;
    if (n < N && k < K) Bt[(size_t)n * K + k] = (half_t)t[lx][yy];
  }
}

// ---------------------------------------------------------------------------
// Per-(b,n) stats: mean / std over S (biased var, +1e-5)
// ---------------------------------------------------------------------------
__global__ __launch_bounds__(128) void k_stats(const float* __restrict__ x,
                                               float* __restrict__ mean,
                                               float* __restrict__ stdv) {
  int bn = blockIdx.x, b = bn >> 3, n = bn & 7;
  int tid = threadIdx.x;
  float s = 0.f, s2 = 0.f;
  for (int t = tid; t < Sc; t += 128) {
    float v = x[((size_t)b * Sc + t) * Nc + n];
    s += v; s2 += v * v;
  }
  __shared__ float r1[128], r2[128];
  r1[tid] = s; r2[tid] = s2; __syncthreads();
  for (int o = 64; o > 0; o >>= 1) {
    if (tid < o) { r1[tid] += r1[tid + o]; r2[tid] += r2[tid + o]; }
    __syncthreads();
  }
  if (tid == 0) {
    float m = r1[0] / (float)Sc;
    float var = fmaxf(r2[0] / (float)Sc - m * m, 0.f);
    mean[bn] = m;
    stdv[bn] = sqrtf(var + 1e-5f);
  }
}

// ---------------------------------------------------------------------------
// Projector conv1d (circular over N, kernel 3, weights [S,3]) + concat stats.
// ---------------------------------------------------------------------------
__global__ __launch_bounds__(128) void k_convproj(const float* __restrict__ x,
                                                  const float* __restrict__ w,
                                                  const float* __restrict__ stat,
                                                  float* __restrict__ z) {
  int bn = blockIdx.x, b = bn >> 3, n = bn & 7;
  int tid = threadIdx.x;
  float s = 0.f;
  for (int i = tid; i < Sc * 3; i += 128) {
    int srow = i / 3, kk = i - srow * 3;
    int nn = (n - 1 + kk + Nc) & 7;
    s += x[((size_t)b * Sc + srow) * Nc + nn] * w[srow * 3 + kk];
  }
  __shared__ float r1[128];
  r1[tid] = s; __syncthreads();
  for (int o = 64; o > 0; o >>= 1) {
    if (tid < o) r1[tid] += r1[tid + o];
    __syncthreads();
  }
  if (tid == 0) {
    z[b * 16 + n] = r1[0];
    z[b * 16 + 8 + n] = stat[bn];
  }
}

// Small dense: out[rows,O] = act(in[rows,K] @ W[K,O] + b). 1 thread / output.
__global__ __launch_bounds__(256) void k_dense_small(
    const float* __restrict__ in, const float* __restrict__ W,
    const float* __restrict__ bias, float* __restrict__ out,
    int rows, int K, int O, int act /*0 none,1 relu,2 exp*/) {
  int i = blockIdx.x * 256 + threadIdx.x;
  if (i >= rows * O) return;
  int r = i / O, o = i - r * O;
  float s = bias ? bias[o] : 0.f;
  for (int t = 0; t < K; ++t) s += in[r * K + t] * W[t * O + o];
  if (act == 1) s = fmaxf(s, 0.f);
  else if (act == 2) s = __expf(s);
  out[i] = s;
}

// ---------------------------------------------------------------------------
// Embedding: h[bn,s,d] = xn * emb_w[d] + emb_b[d] + w_pos[s,d]; writes f32+f16
// ---------------------------------------------------------------------------
__global__ __launch_bounds__(256) void k_embed(
    const float* __restrict__ x, const float* __restrict__ mean,
    const float* __restrict__ stdv, const float* __restrict__ emb_w,
    const float* __restrict__ emb_b, const float* __restrict__ w_pos,
    float* __restrict__ h, half_t* __restrict__ h16) {
  size_t i = (size_t)blockIdx.x * 256 + threadIdx.x;
  if (i >= (size_t)Mrows * Dc) return;
  int d = (int)(i & (Dc - 1));
  size_t row = i >> 9;
  int s = (int)(row % Sc);
  int bn = (int)(row / Sc);
  int b = bn >> 3, n = bn & 7;
  float xv = x[((size_t)b * Sc + s) * Nc + n];
  float xn = (xv - mean[bn]) / stdv[bn];
  float v = xn * emb_w[d] + emb_b[d] + w_pos[(size_t)s * Dc + d];
  h[i] = v;
  h16[i] = (half_t)v;
}

// ---------------------------------------------------------------------------
// ProbSparse measure + top-30 per (bn,h). Sample indices from fixed hash
// (reference uses fixed host-side numpy constants).
// ---------------------------------------------------------------------------
__device__ inline unsigned mixhash(unsigned seed, unsigned l, unsigned j) {
  unsigned x = seed * 0x9E3779B9u + l * 0x85EBCA6Bu + j * 0xC2B2AE35u + 0x165667B1u;
  x ^= x >> 16; x *= 0x7FEB352Du; x ^= x >> 15; x *= 0x846CA68Bu; x ^= x >> 16;
  return x;
}

__global__ __launch_bounds__(128) void k_prob_measure(
    const float* __restrict__ q, const float* __restrict__ k,
    int* __restrict__ top_idx, unsigned seed) {
  int bh = blockIdx.x;                // bn*8 + h
  int bn = bh >> 3, h = bh & 7;
  const float* qb = q + (size_t)bn * Sc * Dc + h * DHc;
  const float* kb = k + (size_t)bn * Sc * Dc + h * DHc;
  int tid = threadIdx.x;
  __shared__ float Msh[Sc];
  for (int l = tid; l < Sc; l += 128) {
    float qr[DHc];
    const float* qp = qb + (size_t)l * Dc;
#pragma unroll
    for (int t = 0; t < DHc; ++t) qr[t] = qp[t];
    float mx = -3.0e38f, sm = 0.f;
    for (int j = 0; j < Uc; ++j) {
      unsigned idx = mixhash(seed, (unsigned)l, (unsigned)j) % (unsigned)Sc;
      const float* kr = kb + (size_t)idx * Dc;
      float d = 0.f;
#pragma unroll
      for (int t = 0; t < DHc; ++t) d += qr[t] * kr[t];
      mx = fmaxf(mx, d); sm += d;
    }
    Msh[l] = mx - sm * (1.0f / (float)Sc);
  }
  __syncthreads();
  __shared__ float rv[128];
  __shared__ int ri[128];
  for (int it = 0; it < Uc; ++it) {
    float bv = -3.0e38f; int bi = -1;
    for (int l = tid; l < Sc; l += 128)
      if (Msh[l] > bv) { bv = Msh[l]; bi = l; }
    rv[tid] = bv; ri[tid] = bi; __syncthreads();
    for (int o = 64; o > 0; o >>= 1) {
      if (tid < o && rv[tid + o] > rv[tid]) { rv[tid] = rv[tid + o]; ri[tid] = ri[tid + o]; }
      __syncthreads();
    }
    if (tid == 0) { top_idx[bh * 32 + it] = ri[0]; Msh[ri[0]] = -3.0e38f; }
    __syncthreads();
  }
  if (tid < 2) top_idx[bh * 32 + Uc + tid] = -1;
}

// ---------------------------------------------------------------------------
// Fused ProbSparse attention per (bn,h): WMMA scores -> softmax -> WMMA update.
// ctx (f16 output) = mean(V) broadcast; top rows replaced by update.
// ---------------------------------------------------------------------------
#define SCW 344                  // score row stride (floats)
#define ATW (SCW * 2)            // attn row stride (halfs), overlays score LDS

__global__ __launch_bounds__(128) void k_prob_attn(
    const float* __restrict__ q, const float* __restrict__ k,
    const float* __restrict__ v, const float* __restrict__ tau,
    const float* __restrict__ delta, const int* __restrict__ top_idx,
    half_t* __restrict__ ctx) {
  int bh = blockIdx.x;
  int bn = bh >> 3, h = bh & 7;
  const float* qb = q + (size_t)bn * Sc * Dc + h * DHc;
  const float* kb = k + (size_t)bn * Sc * Dc + h * DHc;
  const float* vb = v + (size_t)bn * Sc * Dc + h * DHc;
  half_t* cb = ctx + (size_t)bn * Sc * Dc + h * DHc;

  __shared__ __attribute__((aligned(16))) float sc[32][SCW];  // scores / attn
  __shared__ __attribute__((aligned(16))) half_t qs[32][72];  // top-query tile
  __shared__ int tix[32];
  __shared__ float vmean[DHc];
  __shared__ unsigned char istop[Sc];

  half_t* at = (half_t*)&sc[0][0];       // attn f16, stride ATW halfs / row

  const int tid = threadIdx.x;
  const int lane = tid & 31;
  const int wave = tid >> 5;

  if (tid < 32) tix[tid] = top_idx[bh * 32 + tid];
  for (int i = tid; i < Sc; i += 128) istop[i] = 0;
  __syncthreads();
  if (tid < 32 && tix[tid] >= 0) istop[tix[tid]] = 1;

  // Q_top -> f16 LDS tile (invalid rows zeroed)
  for (int i = tid; i < 32 * DHc; i += 128) {
    int r = i >> 6, c = i & 63;
    int l = tix[r];
    qs[r][c] = (half_t)((l >= 0) ? qb[(size_t)l * Dc + c] : 0.f);
  }
  // mean of V over L
  if (tid < DHc) {
    float s = 0.f;
    for (int l = 0; l < Sc; ++l) s += vb[(size_t)l * Dc + tid];
    vmean[tid] = s * (1.0f / (float)Sc);
  }
  __syncthreads();

  // ctx initial = mean(V) on non-top rows (top rows written later by scatter)
  for (int i = tid; i < Sc * DHc; i += 128) {
    int l = i >> 6, c = i & 63;
    if (!istop[l]) cb[(size_t)l * Dc + c] = (half_t)vmean[c];
  }

  // Preload Q fragments (2 M-tiles x 2 K-steps)
  v16h aq[2][2];
#pragma unroll
  for (int mi = 0; mi < 2; ++mi)
#pragma unroll
    for (int ks = 0; ks < 2; ++ks)
      aq[mi][ks] = load_afrag(&qs[mi * 16][ks * 32], 72, lane);

  // scores[32 x 336] = Q_top @ K^T, scaled: (s*tau + delta)/sqrt(Dh)
  const float tauv = tau[bn];
  const float* dl = delta + (size_t)bn * Sc;
  const float scale = 0.125f;  // 1/sqrt(64)
  for (int t = wave; t < Sc / 16; t += 4) {
    v8f c0 = {}, c1 = {};
    int col = t * 16 + (lane & 15);
#pragma unroll
    for (int ks = 0; ks < 2; ++ks) {
      const float* kr = kb + (size_t)col * Dc + ks * 32 + (lane >> 4) * 16;
      v16h bf;
#pragma unroll
      for (int j = 0; j < 16; ++j) bf[j] = (half_t)kr[j];
      c0 = wmma_f16(aq[0][ks], bf, c0);
      c1 = wmma_f16(aq[1][ks], bf, c1);
    }
    float dv = dl[col];
    int rb = (lane >> 4) * 8;
#pragma unroll
    for (int r = 0; r < 8; ++r) {
      sc[rb + r][col] = (c0[r] * tauv + dv) * scale;
      sc[16 + rb + r][col] = (c1[r] * tauv + dv) * scale;
    }
  }
  __syncthreads();

  // Row softmax over 336 cols -> f16 attn (K padded to 352), overlaying LDS
  for (int r = wave; r < 32; r += 4) {
    float vals[11];
    float mx = -3.0e38f;
#pragma unroll
    for (int i = 0; i < 11; ++i) {
      int c = lane + i * 32;
      vals[i] = (c < Sc) ? sc[r][c] : -3.0e38f;
      mx = fmaxf(mx, vals[i]);
    }
    for (int o = 16; o > 0; o >>= 1) mx = fmaxf(mx, __shfl_xor(mx, o, 32));
    float sm = 0.f;
#pragma unroll
    for (int i = 0; i < 11; ++i) {
      int c = lane + i * 32;
      vals[i] = (c < Sc) ? __expf(vals[i] - mx) : 0.f;
      sm += vals[i];
    }
    for (int o = 16; o > 0; o >>= 1) sm += __shfl_xor(sm, o, 32);
    float inv = 1.0f / sm;
#pragma unroll
    for (int i = 0; i < 11; ++i) {
      int c = lane + i * 32;
      if (c < 352) at[(size_t)r * ATW + c] = (half_t)(vals[i] * inv);
    }
  }
  __syncthreads();

  // upd[32 x 64] = attn[32 x 352] @ V[352(pad) x 64]; wave owns one 16-col tile
  v8f u0 = {}, u1 = {};
  const int colv = wave * 16 + (lane & 15);
  for (int ks = 0; ks < 11; ++ks) {
    v16h a0 = load_afrag(at + (size_t)0 * ATW + ks * 32, ATW, lane);
    v16h a1 = load_afrag(at + (size_t)16 * ATW + ks * 32, ATW, lane);
    int kbase = ks * 32 + (lane >> 4) * 16;
    v16h bf;
#pragma unroll
    for (int j = 0; j < 16; ++j) {
      int kk = kbase + j;
      bf[j] = (kk < Sc) ? (half_t)vb[(size_t)kk * Dc + colv] : (half_t)0.f;
    }
    u0 = wmma_f16(a0, bf, u0);
    u1 = wmma_f16(a1, bf, u1);
  }
  // scatter top rows into ctx (f16)
  int rb = (lane >> 4) * 8;
#pragma unroll
  for (int r = 0; r < 8; ++r) {
    int l0 = tix[rb + r];
    if (l0 >= 0) cb[(size_t)l0 * Dc + colv] = (half_t)u0[r];
    int l1 = tix[16 + rb + r];
    if (l1 >= 0) cb[(size_t)l1 * Dc + colv] = (half_t)u1[r];
  }
}

// ---------------------------------------------------------------------------
// out = LN(x (+ rsd)) * g + b over D=512; writes f32 and f16 mirrors. In-place
// safe (row fully register-cached before stores).
// ---------------------------------------------------------------------------
__global__ __launch_bounds__(128) void k_add_ln(
    const float* __restrict__ x, const float* __restrict__ rsd,
    const float* __restrict__ g, const float* __restrict__ b,
    float* __restrict__ out, half_t* __restrict__ out16) {
  int row = blockIdx.x, tid = threadIdx.x;
  const float* xr = x + (size_t)row * Dc;
  const float* rr = rsd ? rsd + (size_t)row * Dc : nullptr;
  float vals[4];
  float s = 0.f;
#pragma unroll
  for (int i = 0; i < 4; ++i) {
    int c = tid + i * 128;
    float vv = xr[c] + (rr ? rr[c] : 0.f);
    vals[i] = vv; s += vv;
  }
  __shared__ float r1[128];
  r1[tid] = s; __syncthreads();
  for (int o = 64; o > 0; o >>= 1) { if (tid < o) r1[tid] += r1[tid + o]; __syncthreads(); }
  float mean = r1[0] / (float)Dc;
  __syncthreads();
  float vs = 0.f;
#pragma unroll
  for (int i = 0; i < 4; ++i) { float d = vals[i] - mean; vs += d * d; }
  r1[tid] = vs; __syncthreads();
  for (int o = 64; o > 0; o >>= 1) { if (tid < o) r1[tid] += r1[tid + o]; __syncthreads(); }
  float inv = rsqrtf(r1[0] / (float)Dc + 1e-5f);
  __syncthreads();
#pragma unroll
  for (int i = 0; i < 4; ++i) {
    int c = tid + i * 128;
    float v = (vals[i] - mean) * inv * g[c] + b[c];
    out[(size_t)row * Dc + c] = v;
    out16[(size_t)row * Dc + c] = (half_t)v;
  }
}

// out[b,p,n] = head_tmp[bn,p] * std[bn] + mean[bn]
__global__ __launch_bounds__(256) void k_head_fin(
    const float* __restrict__ tmp, const float* __restrict__ mean,
    const float* __restrict__ stdv, float* __restrict__ out) {
  int i = blockIdx.x * 256 + threadIdx.x;
  if (i >= Bc * Pc * Nc) return;
  int n = i & 7;
  int p = (i >> 3) % Pc;
  int b = i / (Pc * Nc);
  int bn = b * Nc + n;
  out[i] = tmp[(size_t)bn * Pc + p] * stdv[bn] + mean[bn];
}

// ---------------------------------------------------------------------------
extern "C" void kernel_launch(void* const* d_in, const int* in_sizes, int n_in,
                              void* d_out, int out_size, void* d_ws, size_t ws_size,
                              hipStream_t stream) {
  (void)in_sizes; (void)n_in; (void)out_size; (void)ws_size;
  const float* x_enc   = (const float*)d_in[0];
  const float* emb_w   = (const float*)d_in[4];
  const float* emb_b   = (const float*)d_in[5];
  const float* w_pos   = (const float*)d_in[6];
  const float* tau_conv= (const float*)d_in[7];
  const float* tau_w1  = (const float*)d_in[8];
  const float* tau_b1  = (const float*)d_in[9];
  const float* tau_w2  = (const float*)d_in[10];
  const float* tau_b2  = (const float*)d_in[11];
  const float* tau_w3  = (const float*)d_in[12];
  const float* del_conv= (const float*)d_in[13];
  const float* del_w1  = (const float*)d_in[14];
  const float* del_b1  = (const float*)d_in[15];
  const float* del_w2  = (const float*)d_in[16];
  const float* del_b2  = (const float*)d_in[17];
  const float* del_w3  = (const float*)d_in[18];
  const float* wq      = (const float*)d_in[19];
  const float* bq      = (const float*)d_in[20];
  const float* wk      = (const float*)d_in[21];
  const float* bk      = (const float*)d_in[22];
  const float* wv      = (const float*)d_in[23];
  const float* bv      = (const float*)d_in[24];
  const float* wo      = (const float*)d_in[25];
  const float* bo      = (const float*)d_in[26];
  const float* ffn_w1  = (const float*)d_in[27];
  const float* ffn_b1  = (const float*)d_in[28];
  const float* ffn_w2  = (const float*)d_in[29];
  const float* ffn_b2  = (const float*)d_in[30];
  const float* ln1_g   = (const float*)d_in[31];
  const float* ln1_b   = (const float*)d_in[32];
  const float* ln2_g   = (const float*)d_in[33];
  const float* ln2_b   = (const float*)d_in[34];
  const float* lnf_g   = (const float*)d_in[35];
  const float* lnf_b   = (const float*)d_in[36];
  const float* head_w  = (const float*)d_in[37];
  const float* head_b  = (const float*)d_in[38];
  float* out = (float*)d_out;

  // ---- workspace layout (bytes) ----
  char* base = (char*)d_ws;
  const size_t SZH = (size_t)Mrows * Dc;            // 11,010,048 elements
  const size_t BT_HALFS = (size_t)172032 * Pc;      // largest Bt (head weight)
  float*  rQ     = (float*)base;                    // f32 [Mrows,512]
  float*  rK     = rQ + SZH;
  float*  rV     = rK + SZH;
  half_t* rT1_16 = (half_t*)base;                   // f16 [Mrows,2048], aliases Q/K/V
  float*  rA     = (float*)(base + 3 * SZH * 4);    // f32 [Mrows,512]
  float*  rH     = (float*)(base + 4 * SZH * 4);    // f32 [Mrows,512]
  half_t* rH16   = (half_t*)(base + 5 * SZH * 4);   // f16 mirror of rH
  half_t* rCtx16 = (half_t*)(base + 5 * SZH * 4 + SZH * 2);
  half_t* Bt16   = (half_t*)(base + 5 * SZH * 4 + 2 * SZH * 2);
  float*  p      = (float*)(base + 5 * SZH * 4 + 2 * SZH * 2 + BT_HALFS * 2);
  float* d_mean  = p; p += 64;
  float* d_std   = p; p += 64;
  float* d_tau   = p; p += 64;
  float* d_delta = p; p += (size_t)Bc * Sc * Nc;    // [BN*S] flat
  float* z0 = p; p += Bc * 16;
  float* z1 = p; p += Bc * HIDc;
  float* z2 = p; p += Bc * HIDc;
  float* d_headtmp = p; p += (size_t)BNc * Pc;
  int* d_top = (int*)p;                             // [512*32]

  // 1) per-series stats
  k_stats<<<BNc, 128, 0, stream>>>(x_enc, d_mean, d_std);

  // 2) tau projector
  k_convproj<<<BNc, 128, 0, stream>>>(x_enc, tau_conv, d_std, z0);
  k_dense_small<<<(Bc * HIDc + 255) / 256, 256, 0, stream>>>(z0, tau_w1, tau_b1, z1, Bc, 16, HIDc, 1);
  k_dense_small<<<(Bc * HIDc + 255) / 256, 256, 0, stream>>>(z1, tau_w2, tau_b2, z2, Bc, HIDc, HIDc, 1);
  k_dense_small<<<1, 256, 0, stream>>>(z2, tau_w3, nullptr, d_tau, Bc, HIDc, Nc, 2);

  // 3) delta projector
  k_convproj<<<BNc, 128, 0, stream>>>(x_enc, del_conv, d_mean, z0);
  k_dense_small<<<(Bc * HIDc + 255) / 256, 256, 0, stream>>>(z0, del_w1, del_b1, z1, Bc, 16, HIDc, 1);
  k_dense_small<<<(Bc * HIDc + 255) / 256, 256, 0, stream>>>(z1, del_w2, del_b2, z2, Bc, HIDc, HIDc, 1);
  k_dense_small<<<(Bc * Sc * Nc + 255) / 256, 256, 0, stream>>>(z2, del_w3, nullptr, d_delta, Bc, HIDc, Sc * Nc, 0);

  // 4) embedding (f32 + f16)
  k_embed<<<(unsigned)((SZH + 255) / 256), 256, 0, stream>>>(
      x_enc, d_mean, d_std, emb_w, emb_b, w_pos, rH, rH16);

  // 5) encoder layers
  const dim3 gQ(Dc / 128, Mrows / 128);     // (4,168)
  const dim3 gF1(DFFc / 128, Mrows / 128);  // (16,168)
  const dim3 tDD(Dc / 32, Dc / 32);         // 512x512 transpose
  for (int layer = 0; layer < 2; ++layer) {
    const float* Wq = wq + (size_t)layer * Dc * Dc;
    const float* Wk = wk + (size_t)layer * Dc * Dc;
    const float* Wv = wv + (size_t)layer * Dc * Dc;
    const float* Wo = wo + (size_t)layer * Dc * Dc;
    const float* Bq = bq + (size_t)layer * Dc;
    const float* Bk = bk + (size_t)layer * Dc;
    const float* Bv = bv + (size_t)layer * Dc;
    const float* Bo = bo + (size_t)layer * Dc;
    const float* W1 = ffn_w1 + (size_t)layer * Dc * DFFc;
    const float* B1 = ffn_b1 + (size_t)layer * DFFc;
    const float* W2 = ffn_w2 + (size_t)layer * DFFc * Dc;
    const float* B2 = ffn_b2 + (size_t)layer * Dc;

    k_transpose_cvt<<<tDD, 256, 0, stream>>>(Wq, Bt16, Dc, Dc);
    gemm_wmma16<<<gQ, 256, 0, stream>>>(rH16, Bt16, Bq, rQ, Mrows, Dc, Dc, MB_BIAS);
    k_transpose_cvt<<<tDD, 256, 0, stream>>>(Wk, Bt16, Dc, Dc);
    gemm_wmma16<<<gQ, 256, 0, stream>>>(rH16, Bt16, Bk, rK, Mrows, Dc, Dc, MB_BIAS);
    k_transpose_cvt<<<tDD, 256, 0, stream>>>(Wv, Bt16, Dc, Dc);
    gemm_wmma16<<<gQ, 256, 0, stream>>>(rH16, Bt16, Bv, rV, Mrows, Dc, Dc, MB_BIAS);

    unsigned seed = 0x2545F491u + (unsigned)layer * 0x9E3779B9u;
    k_prob_measure<<<BNc * Hc, 128, 0, stream>>>(rQ, rK, d_top, seed);
    k_prob_attn<<<BNc * Hc, 128, 0, stream>>>(rQ, rK, rV, d_tau, d_delta, d_top, rCtx16);

    k_transpose_cvt<<<tDD, 256, 0, stream>>>(Wo, Bt16, Dc, Dc);
    gemm_wmma16<<<gQ, 256, 0, stream>>>(rCtx16, Bt16, Bo, rA, Mrows, Dc, Dc, MB_BIAS);
    k_add_ln<<<Mrows, 128, 0, stream>>>(rH, rA, ln1_g + layer * Dc, ln1_b + layer * Dc, rH, rH16);

    k_transpose_cvt<<<dim3(Dc / 32, DFFc / 32), 256, 0, stream>>>(W1, Bt16, Dc, DFFc);
    gemm_wmma16<<<gF1, 256, 0, stream>>>(rH16, Bt16, B1, rT1_16, Mrows, DFFc, Dc,
                                         MB_BIAS | MB_GELU | MB_F16);
    k_transpose_cvt<<<dim3(DFFc / 32, Dc / 32), 256, 0, stream>>>(W2, Bt16, DFFc, Dc);
    gemm_wmma16<<<gQ, 256, 0, stream>>>(rT1_16, Bt16, B2, rA, Mrows, Dc, DFFc, MB_BIAS);
    k_add_ln<<<Mrows, 128, 0, stream>>>(rH, rA, ln2_g + layer * Dc, ln2_b + layer * Dc, rH, rH16);
  }

  // 6) final LN
  k_add_ln<<<Mrows, 128, 0, stream>>>(rH, nullptr, lnf_g, lnf_b, rH, rH16);

  // 7) head: [64, S*D] @ [S*D, 96] + b  (A = rH16 viewed as [64, 172032])
  k_transpose_cvt<<<dim3((172032 + 31) / 32, (Pc + 31) / 32), 256, 0, stream>>>(
      head_w, Bt16, Sc * Dc, Pc);
  gemm_wmma16<<<dim3(1, 1), 256, 0, stream>>>(rH16, Bt16, head_b, d_headtmp,
                                              BNc, Pc, Sc * Dc, MB_BIAS);

  // 8) transpose + de-normalize -> [B, P, N]
  k_head_fin<<<(Bc * Pc * Nc + 255) / 256, 256, 0, stream>>>(d_headtmp, d_mean, d_std, out);
}
